// Encoder_4776003633446
// MI455X (gfx1250) — compile-verified, matrix operand order checked by reading, not run on hardware
//
#include <hip/hip_runtime.h>
#include <cstdint>
#include <cstddef>

// ---- problem constants ----
#define B_   2
#define S_   1024
#define D_   1024
#define H_   16
#define DK_  64
#define DV_  64
#define F_   4096
#define L_   4
#define NTOK (B_ * S_)   // 2048
#define TLD  72          // LDS row stride (bf16 elems), padded vs 64 to spread banks

typedef unsigned short ushort_t;
typedef __attribute__((ext_vector_type(16))) __bf16 v16bf;
typedef __attribute__((ext_vector_type(8)))  float  v8f;

union FragCvt { v16bf bf; uint4 q[2]; };
union Chunk8  { uint4 q; ushort_t u[8]; };

__device__ __forceinline__ ushort_t f2bf(float f) {
  unsigned u = __float_as_uint(f);
  unsigned r = u + 0x7fffu + ((u >> 16) & 1u);   // round-to-nearest-even
  return (ushort_t)(r >> 16);
}

__device__ __forceinline__ v8f vzero() {
  v8f z;
  #pragma unroll
  for (int i = 0; i < 8; ++i) z[i] = 0.0f;
  return z;
}

__device__ __forceinline__ v8f wmma_bf16(v16bf a, v16bf b, v8f c) {
  return __builtin_amdgcn_wmma_f32_16x16x32_bf16(false, a, false, b, (short)0, c, false, false);
}

// Build a 16x32 (bf16) A/B fragment. Per ISA layout: lane<16 -> row = lane,
// K = ks + {0..7, 16..23}; lane>=16 -> same row set, K = ks + {8..15, 24..31}.
// Requires K-major storage -> two contiguous 16B loads (ds_load_b128).
__device__ __forceinline__ v16bf make_frag(const ushort_t* base, size_t row0,
                                           size_t stride, int ks) {
  int lane = threadIdx.x & 31;
  int half = lane >> 4;
  int idx  = lane & 15;
  const ushort_t* p = base + (row0 + (size_t)idx) * stride + (size_t)(ks + 8 * half);
  FragCvt f;
  f.q[0] = *(const uint4*)p;
  f.q[1] = *(const uint4*)(p + 16);
  return f.bf;
}

// CDNA5 async copy: 16 bytes global -> LDS, tracked by ASYNCcnt (no VGPR data path).
__device__ __forceinline__ void async_load16(unsigned lds_byte_off, const ushort_t* gptr) {
  asm volatile("global_load_async_to_lds_b128 %0, %1, off"
               :: "v"(lds_byte_off), "v"(gptr)
               : "memory");
}
__device__ __forceinline__ void wait_async0() {
  asm volatile("s_wait_asynccnt 0x0" ::: "memory");
}

// Issue async loads for one 128x64 bf16 tile into LDS (stride TLD).
__device__ __forceinline__ void async_tile_128x64(const ushort_t* g, size_t row0,
                                                  size_t ld, size_t col0, int t,
                                                  unsigned lds_base_byte) {
  #pragma unroll
  for (int i = 0; i < 4; ++i) {
    int c = t + (i << 8);            // 1024 chunks of 8 bf16
    int r = c >> 3, c8 = c & 7;
    const ushort_t* gp = g + (row0 + (size_t)r) * ld + col0 + (size_t)(c8 << 3);
    unsigned loff = lds_base_byte + (unsigned)((r * TLD + (c8 << 3)) * 2);
    async_load16(loff, gp);
  }
}

// ---------------- weight convert + transpose (f32 [l][K][N] -> bf16 [l][N][K]) ----
__global__ __launch_bounds__(256)
void convert_transpose_kernel(const float* __restrict__ in, ushort_t* __restrict__ out,
                              int K, int N, size_t total) {
  size_t idx = (size_t)blockIdx.x * 256 + threadIdx.x;
  if (idx >= total) return;
  size_t plane = (size_t)K * N;
  size_t l = idx / plane;
  size_t rem = idx - l * plane;
  size_t k = rem / N;
  size_t n = rem - k * N;
  out[l * plane + n * (size_t)K + k] = f2bf(in[idx]);
}

// ---------------- embedding + sinusoidal posenc ----------------
__global__ __launch_bounds__(256)
void embed_kernel(const int* __restrict__ tokens, const float* __restrict__ emb,
                  float* __restrict__ xf, ushort_t* __restrict__ xb) {
  int row = blockIdx.x;            // 0..NTOK-1
  int t   = threadIdx.x;
  int tok = tokens[row];
  int pos = row & (S_ - 1);
  #pragma unroll
  for (int i = 0; i < 4; ++i) {
    int d = t + (i << 8);
    int i2 = d >> 1;
    float freq = __expf(-(float)(2 * i2) * (9.210340371976184f / (float)D_)); // ln(1e4)/D
    float a  = (float)pos * freq;
    float pe = (d & 1) ? __cosf(a) : __sinf(a);
    float v  = emb[(size_t)tok * D_ + d] + pe;
    xf[(size_t)row * D_ + d] = v;
    xb[(size_t)row * D_ + d] = f2bf(v);
  }
}

// ---------------- bf16 WMMA GEMM: C = A(MxK) * Bt(NxK)^T ----------------
// LDS double-buffered via async global->LDS copies (ASYNCcnt pipeline).
// Dynamic shared memory layout: [ A0 | A1 | B0 | B1 ], each 128*TLD bf16.
// NOTE: all LDS buffer pointers/offsets are computed as scalar runtime values
// (no constant aggregates from the LDS base -> avoids addrspacecast-in-static-
// initializer link failure).
__global__ __launch_bounds__(256, 1)
void gemm_bf16_kernel(const ushort_t* __restrict__ A, const ushort_t* __restrict__ Bt,
                      const float* __restrict__ bias, const float* __restrict__ residual,
                      float* __restrict__ outF, ushort_t* __restrict__ outB,
                      int M, int N, int K, int act) {
  extern __shared__ ushort_t smem[];
  const unsigned tileB   = 128u * TLD * 2u;              // tile size in bytes
  const unsigned smemBase = (unsigned)(uintptr_t)smem;   // LDS byte offset (scalar)

  const int t    = threadIdx.x;
  const int lane = t & 31;
  const int wv   = t >> 5;             // 8 waves
  const int half = lane >> 4, idx = lane & 15;
  const int mG   = wv >> 1;            // 0..3 : 32-row group
  const int nG   = wv & 1;             // 0..1 : 64-col group
  const size_t rowBase = (size_t)blockIdx.y * 128;
  const size_t colBase = (size_t)blockIdx.x * 128;

  v8f acc[2][4];
  #pragma unroll
  for (int mi = 0; mi < 2; ++mi)
    #pragma unroll
    for (int ni = 0; ni < 4; ++ni) acc[mi][ni] = vzero();

  const int KT = K >> 6;               // 64-wide K slabs

  // prologue: slab 0 -> buffers A0 (slot 0) and B0 (slot 2)
  async_tile_128x64(A,  rowBase, (size_t)K, 0, t, smemBase + 0u * tileB);
  async_tile_128x64(Bt, colBase, (size_t)K, 0, t, smemBase + 2u * tileB);
  wait_async0();
  __syncthreads();

  for (int kt = 0; kt < KT; ++kt) {
    const unsigned cur = (unsigned)(kt & 1);
    const bool more = (kt + 1 < KT);
    if (more) {
      const unsigned nxt = cur ^ 1u;
      size_t col0 = (size_t)(kt + 1) << 6;
      async_tile_128x64(A,  rowBase, (size_t)K, col0, t, smemBase + nxt * tileB);
      async_tile_128x64(Bt, colBase, (size_t)K, col0, t, smemBase + (2u + nxt) * tileB);
    }
    const ushort_t* lAc = smem + (size_t)cur * (128 * TLD);
    const ushort_t* lBc = smem + (size_t)(2u + cur) * (128 * TLD);
    #pragma unroll
    for (int ks = 0; ks < 64; ks += 32) {
      v16bf aF0 = make_frag(lAc, mG * 32 + 0,  TLD, ks);
      v16bf aF1 = make_frag(lAc, mG * 32 + 16, TLD, ks);
      #pragma unroll
      for (int ni = 0; ni < 4; ++ni) {
        v16bf bF = make_frag(lBc, nG * 64 + ni * 16, TLD, ks);
        acc[0][ni] = wmma_bf16(aF0, bF, acc[0][ni]);
        acc[1][ni] = wmma_bf16(aF1, bF, acc[1][ni]);
      }
    }
    if (more) wait_async0();
    __syncthreads();
  }

  // epilogue: C layout -> elem r : row = r + 8*half, col = idx (within 16x16 tile)
  #pragma unroll
  for (int mi = 0; mi < 2; ++mi) {
    #pragma unroll
    for (int ni = 0; ni < 4; ++ni) {
      size_t rB0 = rowBase + mG * 32 + mi * 16;
      size_t col = colBase + nG * 64 + ni * 16 + idx;
      float bv = bias ? bias[col] : 0.0f;
      #pragma unroll
      for (int r = 0; r < 8; ++r) {
        size_t row = rB0 + r + 8 * half;
        float v = acc[mi][ni][r] + bv;
        if (act == 1) v = 0.5f * v * (1.0f + erff(v * 0.70710678118654752f)); // exact GELU
        if (residual) v += residual[row * (size_t)N + col];
        if (outF) outF[row * (size_t)N + col] = v;
        if (outB) outB[row * (size_t)N + col] = f2bf(v);
      }
    }
  }
}
#define GEMM_SMEM_BYTES (4u * 128u * TLD * 2u)   // 73728 B (< 320 KB WGP LDS)

// ---------------- flash attention (one block = 64 query rows of one (b,h)) ----------
__global__ __launch_bounds__(128, 1)
void attention_kernel(const ushort_t* __restrict__ Qb, const ushort_t* __restrict__ Kb,
                      const ushort_t* __restrict__ Vb, const int* __restrict__ tokens,
                      ushort_t* __restrict__ Ctx) {
  __shared__ ushort_t Kl[64 * TLD];          // [key][feat]  (async-filled)
  __shared__ ushort_t Vl[64 * TLD];          // [dv][key]    (transposed, ds stores)
  __shared__ ushort_t Pl[4][16 * TLD];       // per-wave P tile [qrow][key]

  const int t    = threadIdx.x;
  const int lane = t & 31;
  const int wv   = t >> 5;                   // 4 waves, 16 query rows each
  const int half = lane >> 4, idx = lane & 15;

  const int qt = blockIdx.x & 15;            // S/64 = 16
  const int h  = (blockIdx.x >> 4) & 15;
  const int b  = blockIdx.x >> 8;
  const int q0 = qt * 64;
  const size_t strideKV = (size_t)H_ * DK_;  // 1024
  const size_t headOff  = (size_t)h * DK_;
  const float  scale = 0.125f;               // 1/sqrt(64)
  unsigned KlBase;
  {
    const ushort_t* kp = &Kl[0];
    KlBase = (unsigned)(uintptr_t)kp;        // scalar runtime value
  }

  // Q fragments (A-layout, direct from global; rows = q0 + wv*16 + lane&15)
  v16bf qF[2];
  qF[0] = make_frag(Qb + headOff, (size_t)(b * S_ + q0 + wv * 16), strideKV, 0);
  qF[1] = make_frag(Qb + headOff, (size_t)(b * S_ + q0 + wv * 16), strideKV, 32);

  float mRow[8], lRow[8];
  v8f o[4];
  #pragma unroll
  for (int r = 0; r < 8; ++r) { mRow[r] = -1e30f; lRow[r] = 0.0f; }
  #pragma unroll
  for (int ni = 0; ni < 4; ++ni) o[ni] = vzero();

  for (int kb = 0; kb < S_; kb += 64) {
    const ushort_t* Kg = Kb + ((size_t)(b * S_ + kb)) * strideKV + headOff;
    const ushort_t* Vg = Vb + ((size_t)(b * S_ + kb)) * strideKV + headOff;
    // stage K via async global->LDS; V transposed through registers
    #pragma unroll
    for (int i = 0; i < 4; ++i) {
      int c = t + (i << 7);                  // 512 chunks of 8 bf16
      int r = c >> 3, c8 = c & 7;
      async_load16(KlBase + (unsigned)((r * TLD + (c8 << 3)) * 2),
                   Kg + (size_t)r * strideKV + (size_t)(c8 << 3));
      Chunk8 ch;
      ch.q = *(const uint4*)(Vg + (size_t)r * strideKV + (size_t)(c8 << 3));
      #pragma unroll
      for (int e = 0; e < 8; ++e) Vl[(c8 * 8 + e) * TLD + r] = ch.u[e];
    }
    wait_async0();
    __syncthreads();

    // scores: 16x64 per wave = 4 tiles x (2 WMMAs over DK=64)
    v8f sT[4];
    #pragma unroll
    for (int kn = 0; kn < 4; ++kn) {
      sT[kn] = vzero();
      sT[kn] = wmma_bf16(qF[0], make_frag(Kl, kn * 16, TLD, 0),  sT[kn]);
      sT[kn] = wmma_bf16(qF[1], make_frag(Kl, kn * 16, TLD, 32), sT[kn]);
    }
    int tk[4];
    #pragma unroll
    for (int kn = 0; kn < 4; ++kn) tk[kn] = tokens[b * S_ + kb + kn * 16 + idx];

    // online softmax per row (row = r + 8*half; cols spread over 16 lanes x 4 tiles)
    #pragma unroll
    for (int r = 0; r < 8; ++r) {
      float mx = -1e30f;
      #pragma unroll
      for (int kn = 0; kn < 4; ++kn) {
        float s = sT[kn][r] * scale;
        if (tk[kn] == 0) s = -1e9f;            // PAD mask on key tokens
        sT[kn][r] = s;
        mx = fmaxf(mx, s);
      }
      #pragma unroll
      for (int m = 8; m >= 1; m >>= 1) mx = fmaxf(mx, __shfl_xor(mx, m, 16));
      float mN   = fmaxf(mRow[r], mx);
      float corr = __expf(mRow[r] - mN);
      float sum  = 0.0f;
      #pragma unroll
      for (int kn = 0; kn < 4; ++kn) {
        float pv = __expf(sT[kn][r] - mN);
        sT[kn][r] = pv;
        sum += pv;
      }
      #pragma unroll
      for (int m = 8; m >= 1; m >>= 1) sum += __shfl_xor(sum, m, 16);
      lRow[r] = lRow[r] * corr + sum;
      mRow[r] = mN;
      #pragma unroll
      for (int ni = 0; ni < 4; ++ni) o[ni][r] *= corr;
      // stage P (C-layout -> LDS row-major [qrow][key]) for A-fragment reload
      #pragma unroll
      for (int kn = 0; kn < 4; ++kn)
        Pl[wv][(r + 8 * half) * TLD + kn * 16 + idx] = f2bf(sT[kn][r]);
    }

    // O += P * V  (per-wave LDS; DS ops are in-order within a wave)
    v16bf pF0 = make_frag(Pl[wv], 0, TLD, 0);
    v16bf pF1 = make_frag(Pl[wv], 0, TLD, 32);
    #pragma unroll
    for (int ni = 0; ni < 4; ++ni) {
      o[ni] = wmma_bf16(pF0, make_frag(Vl, ni * 16, TLD, 0),  o[ni]);
      o[ni] = wmma_bf16(pF1, make_frag(Vl, ni * 16, TLD, 32), o[ni]);
    }
    __syncthreads();
  }

  // finalize: ctx = O / l  -> bf16 [B,S,H,DV]
  #pragma unroll
  for (int ni = 0; ni < 4; ++ni) {
    #pragma unroll
    for (int r = 0; r < 8; ++r) {
      float val = o[ni][r] / lRow[r];
      size_t row = (size_t)(b * S_ + q0 + wv * 16 + r + 8 * half);
      Ctx[row * strideKV + headOff + ni * 16 + idx] = f2bf(val);
    }
  }
}

// ---------------- layernorm: one block per row (D=1024) -> f32 x + bf16 x ----------
__global__ __launch_bounds__(256)
void layernorm_kernel(const float* __restrict__ y, const float* __restrict__ g,
                      const float* __restrict__ bta, float* __restrict__ xf,
                      ushort_t* __restrict__ xb) {
  __shared__ float red[16];
  const int row = blockIdx.x, t = threadIdx.x;
  const int lane = t & 31, wv = t >> 5;
  const float* yr = y + (size_t)row * D_;
  float v[4], s = 0.0f, s2 = 0.0f;
  #pragma unroll
  for (int i = 0; i < 4; ++i) {
    v[i] = yr[t + (i << 8)];
    s += v[i];
    s2 += v[i] * v[i];
  }
  #pragma unroll
  for (int m = 16; m >= 1; m >>= 1) {
    s  += __shfl_xor(s,  m, 32);
    s2 += __shfl_xor(s2, m, 32);
  }
  if (lane == 0) { red[wv] = s; red[8 + wv] = s2; }
  __syncthreads();
  s = 0.0f; s2 = 0.0f;
  #pragma unroll
  for (int w = 0; w < 8; ++w) { s += red[w]; s2 += red[8 + w]; }
  float mu   = s * (1.0f / D_);
  float var  = s2 * (1.0f / D_) - mu * mu;
  float rstd = rsqrtf(var + 1e-5f);
  #pragma unroll
  for (int i = 0; i < 4; ++i) {
    int c = t + (i << 8);
    float o = (v[i] - mu) * rstd * g[c] + bta[c];
    xf[(size_t)row * D_ + c] = o;
    xb[(size_t)row * D_ + c] = f2bf(o);
  }
}

// ---------------- host orchestration ----------------
extern "C" void kernel_launch(void* const* d_in, const int* in_sizes, int n_in,
                              void* d_out, int out_size, void* d_ws, size_t ws_size,
                              hipStream_t stream) {
  const int*   tokens = (const int*)  d_in[0];
  const float* emb    = (const float*)d_in[1];
  const float* Wq = (const float*)d_in[2];  const float* bq = (const float*)d_in[3];
  const float* Wk = (const float*)d_in[4];  const float* bk = (const float*)d_in[5];
  const float* Wv = (const float*)d_in[6];  const float* bv = (const float*)d_in[7];
  const float* Wo = (const float*)d_in[8];  const float* bo = (const float*)d_in[9];
  const float* ln1g = (const float*)d_in[10]; const float* ln1b = (const float*)d_in[11];
  const float* W1 = (const float*)d_in[12]; const float* W2 = (const float*)d_in[13];
  const float* ln2g = (const float*)d_in[14]; const float* ln2b = (const float*)d_in[15];
  float* x = (float*)d_out;                  // running activations, f32 [B,S,D]

  // workspace carve-up (256B aligned)
  char* p = (char*)d_ws;
  auto alloc = [&](size_t bytes) -> char* {
    char* r = p; p += (bytes + 255) & ~(size_t)255; return r;
  };
  const size_t WD  = (size_t)L_ * D_ * (H_ * DK_);   // 4M elems
  const size_t WF  = (size_t)L_ * D_ * F_;           // 16M elems
  ushort_t* WqT = (ushort_t*)alloc(WD * 2);
  ushort_t* WkT = (ushort_t*)alloc(WD * 2);
  ushort_t* WvT = (ushort_t*)alloc(WD * 2);
  ushort_t* WoT = (ushort_t*)alloc(WD * 2);
  ushort_t* W1T = (ushort_t*)alloc(WF * 2);
  ushort_t* W2T = (ushort_t*)alloc(WF * 2);
  ushort_t* xb   = (ushort_t*)alloc((size_t)NTOK * D_ * 2);
  ushort_t* qbuf = (ushort_t*)alloc((size_t)NTOK * (H_ * DK_) * 2);
  ushort_t* kbuf = (ushort_t*)alloc((size_t)NTOK * (H_ * DK_) * 2);
  ushort_t* vbuf = (ushort_t*)alloc((size_t)NTOK * (H_ * DV_) * 2);
  ushort_t* ctxb = (ushort_t*)alloc((size_t)NTOK * (H_ * DV_) * 2);
  ushort_t* hbuf = (ushort_t*)alloc((size_t)NTOK * F_ * 2);
  float*    ybuf = (float*)   alloc((size_t)NTOK * D_ * 4);

  // weight convert + transpose (weights then live in the 192MB L2)
  convert_transpose_kernel<<<(unsigned)((WD + 255) / 256), 256, 0, stream>>>(Wq, WqT, D_, H_ * DK_, WD);
  convert_transpose_kernel<<<(unsigned)((WD + 255) / 256), 256, 0, stream>>>(Wk, WkT, D_, H_ * DK_, WD);
  convert_transpose_kernel<<<(unsigned)((WD + 255) / 256), 256, 0, stream>>>(Wv, WvT, D_, H_ * DV_, WD);
  convert_transpose_kernel<<<(unsigned)((WD + 255) / 256), 256, 0, stream>>>(Wo, WoT, H_ * DV_, D_, WD);
  convert_transpose_kernel<<<(unsigned)((WF + 255) / 256), 256, 0, stream>>>(W1, W1T, D_, F_, WF);
  convert_transpose_kernel<<<(unsigned)((WF + 255) / 256), 256, 0, stream>>>(W2, W2T, F_, D_, WF);

  embed_kernel<<<NTOK, 256, 0, stream>>>(tokens, emb, x, xb);

  const dim3 gD(D_ / 128, NTOK / 128);   // N=1024 GEMMs
  const dim3 gF(F_ / 128, NTOK / 128);   // N=4096 GEMM (FFN1)

  for (int l = 0; l < L_; ++l) {
    const size_t wOff = (size_t)l * D_ * (H_ * DK_);
    const size_t fOff = (size_t)l * D_ * F_;
    // Q, K, V projections -> bf16
    gemm_bf16_kernel<<<gD, 256, GEMM_SMEM_BYTES, stream>>>(
        xb, WqT + wOff, bq + (size_t)l * (H_ * DK_),
        nullptr, nullptr, qbuf, NTOK, H_ * DK_, D_, 0);
    gemm_bf16_kernel<<<gD, 256, GEMM_SMEM_BYTES, stream>>>(
        xb, WkT + wOff, bk + (size_t)l * (H_ * DK_),
        nullptr, nullptr, kbuf, NTOK, H_ * DK_, D_, 0);
    gemm_bf16_kernel<<<gD, 256, GEMM_SMEM_BYTES, stream>>>(
        xb, WvT + wOff, bv + (size_t)l * (H_ * DV_),
        nullptr, nullptr, vbuf, NTOK, H_ * DV_, D_, 0);
    // attention -> ctx bf16
    attention_kernel<<<B_ * H_ * (S_ / 64), 128, 0, stream>>>(qbuf, kbuf, vbuf, tokens, ctxb);
    // out proj + bias + residual -> y f32 ; LN1 -> x f32 + xb bf16
    gemm_bf16_kernel<<<gD, 256, GEMM_SMEM_BYTES, stream>>>(
        ctxb, WoT + wOff, bo + (size_t)l * D_,
        x, ybuf, nullptr, NTOK, D_, H_ * DV_, 0);
    layernorm_kernel<<<NTOK, 256, 0, stream>>>(ybuf, ln1g + (size_t)l * D_, ln1b + (size_t)l * D_, x, xb);
    // FFN: h = GELU(x W1) bf16 ; y = h W2 + x ; LN2 -> x
    gemm_bf16_kernel<<<gF, 256, GEMM_SMEM_BYTES, stream>>>(
        xb, W1T + fOff, nullptr,
        nullptr, nullptr, hbuf, NTOK, F_, D_, 1);
    gemm_bf16_kernel<<<gD, 256, GEMM_SMEM_BYTES, stream>>>(
        hbuf, W2T + fOff, nullptr,
        x, ybuf, nullptr, NTOK, D_, F_, 0);
    layernorm_kernel<<<NTOK, 256, 0, stream>>>(ybuf, ln2g + (size_t)l * D_, ln2b + (size_t)l * D_, x, xb);
  }
}
